// GraphAttentionLayer_47820165873787
// MI455X (gfx1250) — matrix operand order repaired and checked by reading, not compiled, not run on hardware
//
#include <hip/hip_runtime.h>

typedef __attribute__((ext_vector_type(16))) _Float16 v16h;
typedef __attribute__((ext_vector_type(8)))  float    v8f;
typedef __attribute__((ext_vector_type(4)))  int      v4i;

#define ALPHA   0.2f
#define NEGINF  -9.0e15f
#define B_      4
#define N_      4096
#define FIN     128
#define FOUT    64

union V16H { v16h h; int4 q[2]; unsigned u[8]; };

// ---------------- Kernel 1: h = x@W, plus packed f16 variants + row scores ---
__global__ __launch_bounds__(256) void gat_prep(
    const float* __restrict__ x, const float* __restrict__ W,
    const float* __restrict__ a,
    _Float16* __restrict__ hF,   // [b][n][f] row-major f16
    _Float16* __restrict__ hA,   // [b][n][f] = h*a3, f16
    _Float16* __restrict__ hT,   // [b][f][n] transposed f16
    float* __restrict__ s1g, float* __restrict__ s2g)
{
    __shared__ float Wl[FIN * FOUT];
    __shared__ float al[3 * FOUT];
    const int tid = threadIdx.x;
    for (int i = tid; i < FIN * FOUT; i += 256) Wl[i] = W[i];
    if (tid < 3 * FOUT) al[tid] = a[tid];
    __syncthreads();

    const int g = blockIdx.x * 256 + tid;       // global row 0..16383
    const int b = g >> 12, n = g & (N_ - 1);

    float acc[FOUT];
#pragma unroll
    for (int f = 0; f < FOUT; ++f) acc[f] = 0.f;

    const float* xr = x + (size_t)g * FIN;
    for (int k = 0; k < FIN; ++k) {
        const float xv = xr[k];
        const float* wr = &Wl[k * FOUT];
#pragma unroll
        for (int f = 0; f < FOUT; ++f) acc[f] = fmaf(xv, wr[f], acc[f]);
    }

    float s1 = 0.f, s2 = 0.f;
#pragma unroll
    for (int f = 0; f < FOUT; ++f) {
        s1 = fmaf(acc[f], al[f], s1);
        s2 = fmaf(acc[f], al[64 + f], s2);
    }
    s1g[g] = s1;
    s2g[g] = s2;

    _Float16* hFr = hF + (size_t)g * FOUT;
    _Float16* hAr = hA + (size_t)g * FOUT;
#pragma unroll
    for (int f = 0; f < FOUT; ++f) {
        const _Float16 hv = (_Float16)acc[f];
        hFr[f] = hv;
        hAr[f] = (_Float16)(acc[f] * al[128 + f]);
        hT[((size_t)b * FOUT + f) * N_ + n] = hv;
    }
}

// ---------------- Kernel 2: fused masked-softmax attention (flash style) -----
// One wave per 16-row i-block. Computes transposed score tiles S'[j][i] so the
// softmax over j is per-lane; accumulates O'[f][i] = h^T @ P'.
__global__ __launch_bounds__(256) void gat_main(
    const int* __restrict__ adj,
    const _Float16* __restrict__ hF, const _Float16* __restrict__ hA,
    const _Float16* __restrict__ hT,
    const float* __restrict__ s1g, const float* __restrict__ s2g,
    float* __restrict__ out)
{
    const int lane = threadIdx.x & 31;
    const int wave = threadIdx.x >> 5;
    const int il   = lane & 15;     // column index (i) within tile
    const int hh   = lane >> 4;     // half-wave id
    const int b    = blockIdx.x >> 5;                       // 4 batches
    const int i0   = (blockIdx.x & 31) * 128 + wave * 16;   // row tile base

    // B operands for S' (i-block constant): rows of h*a3 for columns i, K=0..63
    V16H Bs0, Bs1;
    {
        const _Float16* p = hA + (size_t)(b * N_ + i0 + il) * FOUT + hh * 8;
        Bs0.q[0] = *(const int4*)(p);
        Bs0.q[1] = *(const int4*)(p + 16);
        Bs1.q[0] = *(const int4*)(p + 32);
        Bs1.q[1] = *(const int4*)(p + 48);
    }
    const float s1l = s1g[b * N_ + i0 + il];

    float m = -3.0e38f, l = 0.f;
    v8f O[4];
#pragma unroll
    for (int t = 0; t < 4; ++t) O[t] = (v8f){};

    const int*      adjRow = adj + ((size_t)(b * N_) + i0 + il) * N_;
    const float*    s2row  = s2g + b * N_;
    const _Float16* hFb    = hF + (size_t)b * N_ * FOUT;
    const _Float16* hTb    = hT + (size_t)b * FOUT * N_;

#pragma unroll 1
    for (int j0 = 0; j0 < N_; j0 += 32) {
        // S'[j][i] tiles, jm selects j sub-tile of 16; K=64 via two chained WMMA
        v8f S[2];
#pragma unroll
        for (int jm = 0; jm < 2; ++jm) {
            const _Float16* ar = hFb + (size_t)(j0 + jm * 16 + il) * FOUT + hh * 8;
            V16H A0, A1;
            A0.q[0] = *(const int4*)(ar);
            A0.q[1] = *(const int4*)(ar + 16);
            A1.q[0] = *(const int4*)(ar + 32);
            A1.q[1] = *(const int4*)(ar + 48);
            v8f c = (v8f){};
            c = __builtin_amdgcn_wmma_f32_16x16x32_f16(false, A0.h, false, Bs0.h,
                                                       (short)0, c, false, false);
            c = __builtin_amdgcn_wmma_f32_16x16x32_f16(false, A1.h, false, Bs1.h,
                                                       (short)0, c, false, false);
            S[jm] = c;
        }

        // adjacency mask (non-temporal: 268 MB single-use stream, keep L2 for h)
        v4i aj[4];
        {
            const int* ap = adjRow + j0 + hh * 8;
            aj[0] = __builtin_nontemporal_load((const v4i*)(ap));
            aj[1] = __builtin_nontemporal_load((const v4i*)(ap + 4));
            aj[2] = __builtin_nontemporal_load((const v4i*)(ap + 16));
            aj[3] = __builtin_nontemporal_load((const v4i*)(ap + 20));
        }
        float4 s2v[4];
        {
            const float* sp = s2row + j0 + hh * 8;
            s2v[0] = *(const float4*)(sp);
            s2v[1] = *(const float4*)(sp + 4);
            s2v[2] = *(const float4*)(sp + 16);
            s2v[3] = *(const float4*)(sp + 20);
        }

        float e[16];
#pragma unroll
        for (int jm = 0; jm < 2; ++jm)
#pragma unroll
            for (int r = 0; r < 8; ++r) {
                const float s2s = ((const float*)&s2v[jm * 2 + (r >> 2)])[r & 3];
                const int   av  = aj[jm * 2 + (r >> 2)][r & 3];
                float ev = S[jm][r] + s1l + s2s;
                ev = fmaxf(ev, ALPHA * ev);           // leaky relu
                e[jm * 8 + r] = (av > 0) ? ev : NEGINF;
            }

        // online softmax (per-lane column i; merge max across half-wave pair)
        float tm = e[0];
#pragma unroll
        for (int t = 1; t < 16; ++t) tm = fmaxf(tm, e[t]);
        tm = fmaxf(tm, __shfl_xor(tm, 16, 32));
        const float mold = m;
        const float mnew = fmaxf(m, tm);
        const float scale = __expf(mold - mnew);
        m = mnew;

        float p[16], ls = 0.f;
#pragma unroll
        for (int t = 0; t < 16; ++t) { p[t] = __expf(e[t] - mnew); ls += p[t]; }
        l = l * scale + ls;

        // rescale accumulators only when some lane's running max moved
        if (__ballot(mnew > mold) != 0ull) {
#pragma unroll
            for (int t = 0; t < 4; ++t)
#pragma unroll
                for (int r = 0; r < 8; ++r) O[t][r] *= scale;
        }

        // pack P' into WMMA B-operand layout (K = j, N = i)
        unsigned pk0[4], pk1[4];
#pragma unroll
        for (int u = 0; u < 4; ++u) {
            auto t0 = __builtin_amdgcn_cvt_pkrtz(p[2 * u], p[2 * u + 1]);
            auto t1 = __builtin_amdgcn_cvt_pkrtz(p[8 + 2 * u], p[8 + 2 * u + 1]);
            pk0[u] = __builtin_bit_cast(unsigned, t0);
            pk1[u] = __builtin_bit_cast(unsigned, t1);
        }
        V16H Bp;
#pragma unroll
        for (int u = 0; u < 4; ++u) {
            const unsigned o1 = (unsigned)__shfl_xor((int)pk1[u], 16, 32);
            const unsigned o0 = (unsigned)__shfl_xor((int)pk0[u], 16, 32);
            Bp.u[u]     = hh ? o1 : pk0[u];
            Bp.u[4 + u] = hh ? pk1[u] : o0;
        }

        // O'[f][i] += h^T tile (A) x P' (B), 4 feature tiles of 16
#pragma unroll
        for (int ft = 0; ft < 4; ++ft) {
            const _Float16* tr = hTb + (size_t)(ft * 16 + il) * N_ + j0 + hh * 16;
            V16H At;
            At.q[0] = *(const int4*)(tr);
            At.q[1] = *(const int4*)(tr + 8);
            O[ft] = __builtin_amdgcn_wmma_f32_16x16x32_f16(false, At.h, false, Bp.h,
                                                           (short)0, O[ft], false, false);
        }
    }

    // finalize: combine half-wave sums, normalize, ELU, store (NT: written once)
    const float lt  = l + __shfl_xor(l, 16, 32);
    const float inv = 1.0f / lt;
    float* orow = out + ((size_t)(b * N_) + i0 + il) * FOUT + hh * 8;
#pragma unroll
    for (int ft = 0; ft < 4; ++ft)
#pragma unroll
        for (int r = 0; r < 8; ++r) {
            float v = O[ft][r] * inv;
            v = (v > 0.f) ? v : (__expf(v) - 1.0f);   // ELU
            __builtin_nontemporal_store(v, &orow[ft * 16 + r]);
        }
}

// ---------------- launch ----------------------------------------------------
extern "C" void kernel_launch(void* const* d_in, const int* in_sizes, int n_in,
                              void* d_out, int out_size, void* d_ws, size_t ws_size,
                              hipStream_t stream) {
    (void)in_sizes; (void)n_in; (void)out_size; (void)ws_size;
    const float* x   = (const float*)d_in[0];
    const int*   adj = (const int*)d_in[1];
    const float* W   = (const float*)d_in[2];
    const float* a   = (const float*)d_in[3];

    char* ws = (char*)d_ws;
    _Float16* hF = (_Float16*)(ws);                            // 2 MB
    _Float16* hA = (_Float16*)(ws + (2u << 20));               // 2 MB
    _Float16* hT = (_Float16*)(ws + (4u << 20));               // 2 MB
    float*    s1 = (float*)(ws + (6u << 20));                  // 64 KB
    float*    s2 = (float*)(ws + (6u << 20) + (64u << 10));    // 64 KB

    gat_prep<<<64, 256, 0, stream>>>(x, W, a, hF, hA, hT, s1, s2);
    gat_main<<<128, 256, 0, stream>>>(adj, hF, hA, hT, s1, s2, (float*)d_out);
}
